// ThreeBodyInteraction_60919816127131
// MI455X (gfx1250) — matrix operand order repaired and checked by reading, not compiled, not run on hardware
//
#include <hip/hip_runtime.h>

// ---------------------------------------------------------------------------
// MI455X (gfx1250) three-body interaction block.
//   * bf16 WMMA (v_wmma_f32_16x16x32_bf16) for all three GEMMs
//   * TDM gather-mode DMA (tensor_load_to_lds) pulls indexed bf16 edge_attr
//     rows straight into the strided LDS A-tile (pad_enable), overlapped with
//     the angle MLP; completion via s_wait_tensorcnt. Fallback to synchronous
//     VGPR gather when ws_size cannot hold the bf16 edge_attr cache.
//   * segment_sum accumulated into d_out with agent-scope f32 atomics,
//     final delta@Wu done in place on d_out.
// ---------------------------------------------------------------------------

typedef __attribute__((ext_vector_type(16))) __bf16    v16bf;
typedef __attribute__((ext_vector_type(8)))  __bf16    v8bf;
typedef __attribute__((ext_vector_type(8)))  float     v8f;
typedef __attribute__((ext_vector_type(4)))  unsigned  v4u;
typedef __attribute__((ext_vector_type(8)))  unsigned  v8u;

#define DMODEL 128
#define KIN    276          // 2*D + NB
#define KPAD   288          // KIN padded to multiple of 32
#define NBF    20
#define AS_STRIDE 296       // 592B rows: 16B aligned; TDM pad skips to this
#define HS_STRIDE 136       // 272B rows: 16B aligned

static __device__ __forceinline__ float silu_f(float x) {
  return x / (1.0f + __expf(-x));
}

static __device__ __forceinline__ v16bf cat16(v8bf lo, v8bf hi) {
  v16bf r;
#pragma unroll
  for (int i = 0; i < 8; ++i) { r[i] = lo[i]; r[i + 8] = hi[i]; }
  return r;
}

// A-fragment (16x32 bf16, ISA 7.12.2): lane L holds row M=L&15;
// elements 0..7 <-> K=kbase+g*8+e, elements 8..15 <-> K=kbase+16+g*8+e (g=L>>4).
static __device__ __forceinline__ v16bf load_afrag(const __bf16* row, int kbase, int g) {
  const v8bf* lo = (const v8bf*)(row + kbase + g * 8);
  const v8bf* hi = (const v8bf*)(row + kbase + 16 + g * 8);
  return cat16(*lo, *hi);
}

// B-fragment pre-packed: tile = ks*8 + nt holds 512 halves lane-major,
// each lane reads its 16 contiguous halves (32 bytes).
static __device__ __forceinline__ v16bf load_bfrag(const __bf16* wp, int tile, int lane) {
  const v8bf* p = (const v8bf*)(wp + tile * 512 + lane * 16);
  return cat16(p[0], p[1]);
}

// ---- TDM gather: 8 indexed rows of 128 bf16 -> LDS, strided by AS_STRIDE.
// D# per CDNA5 ISA (gather_mode=1, 32-bit indices,
// data_size=1(2B), tile_dim0=128, pad: 64 DW interval (code 5), 84 DW (code 83)).
static __device__ __forceinline__ void tdm_gather8(unsigned lds_addr,
                                                   const void* gbase,
                                                   const int* idx,
                                                   unsigned nrows) {
  unsigned long long ga = (unsigned long long)gbase;
  v4u g0;
  g0[0] = 0xC0000001u;                                   // count=1 | idx32 | gather
  g0[1] = lds_addr;                                      // lds_addr[31:0]
  g0[2] = (unsigned)ga;                                  // global_addr[31:0]
  g0[3] = (unsigned)((ga >> 32) & 0x01FFFFFFu) | 0x80000000u;  // addr hi | type=2
  v8u g1;
  g1[0] = (1u << 16) | (1u << 20) | (5u << 22) | (83u << 25);  // dsz=2B,pad on
  g1[1] = (128u & 0xFFFFu) << 16;                        // tensor_dim0 lo16
  g1[2] = ((nrows & 0xFFFFu) << 16);                     // dim0 hi16=0 | dim1 lo16
  g1[3] = (nrows >> 16) | (128u << 16);                  // dim1 hi16 | tile_dim0
  g1[4] = 8u;                                            // tile_dim1 = #indices
  g1[5] = 128u;                                          // tensor_dim0_stride lo32
  g1[6] = 0u;
  g1[7] = 0u;
  v4u g2, g3;
#pragma unroll
  for (int i = 0; i < 4; ++i) { g2[i] = (unsigned)idx[i]; g3[i] = (unsigned)idx[4 + i]; }
  asm volatile("tensor_load_to_lds %0, %1, %2, %3"
               :: "s"(g0), "s"(g1), "s"(g2), "s"(g3)
               : "memory");
}

static __device__ __forceinline__ void wait_tensorcnt0() {
#if __has_builtin(__builtin_amdgcn_s_wait_tensorcnt)
  __builtin_amdgcn_s_wait_tensorcnt(0);
#else
  asm volatile("s_wait_tensorcnt 0x0" ::: "memory");
#endif
}

// --------------------------- preparation kernels ---------------------------

__global__ void zero_f32_kernel(float* __restrict__ p, long long n) {
  long long i = (long long)blockIdx.x * blockDim.x + threadIdx.x;
  long long stride = (long long)gridDim.x * blockDim.x;
  for (; i < n; i += stride) p[i] = 0.0f;
}

__global__ void cvt_bf16_kernel(const float* __restrict__ src,
                                __bf16* __restrict__ dst, long long n) {
  long long i = (long long)blockIdx.x * blockDim.x + threadIdx.x;
  long long stride = (long long)gridDim.x * blockDim.x;
  for (; i < n; i += stride) dst[i] = (__bf16)src[i];
}

// Per-edge geometry: len = max(||v||, eps), unit vector v/len.
__global__ void geom_kernel(const float* __restrict__ ev,
                            float4* __restrict__ geom, int E) {
  int e = blockIdx.x * blockDim.x + threadIdx.x;
  if (e >= E) return;
  float x = ev[e * 3 + 0], y = ev[e * 3 + 1], z = ev[e * 3 + 2];
  float len = fmaxf(sqrtf(x * x + y * y + z * z), 1e-6f);
  float inv = 1.0f / len;
  geom[e] = make_float4(len, x * inv, y * inv, z * inv);
}

// Pack KxN (row-major, N=128) f32 weight into bf16 WMMA B-fragment order.
__global__ void pack_b_kernel(const float* __restrict__ W, __bf16* __restrict__ out,
                              int K, int Kpad) {
  int i = blockIdx.x * blockDim.x + threadIdx.x;
  int total = (Kpad / 32) * 8 * 512;
  if (i >= total) return;
  int tile = i >> 9;
  int lane = (i >> 4) & 31;
  int e    = i & 15;
  int ks = tile >> 3, nt = tile & 7;
  int k = ks * 32 + (lane >> 4) * 8 + (e & 7) + (e >> 3) * 16;
  int n = nt * 16 + (lane & 15);
  float v = (k < K) ? W[k * DMODEL + n] : 0.0f;
  out[i] = (__bf16)v;
}

// -------------------- shared pieces of the triplet stage -------------------

static __device__ __forceinline__ void angle_row(
    int r, const int* es, const int* ek, const float4* __restrict__ geom,
    const float* __restrict__ Wa1, const float* __restrict__ ba1,
    const float* __restrict__ Wa2, const float* __restrict__ ba2,
    __bf16 (&As)[16][AS_STRIDE]) {
  float af[NBF];
  if (es[r] >= 0) {
    float4 gi = geom[es[r]];
    float4 gk = geom[ek[r]];
    float cosv = gi.y * gk.y + gi.z * gk.z + gi.w * gk.w;
    cosv = fminf(fmaxf(cosv, -1.0f), 1.0f);
    float a0 = gi.x, a1 = gk.x, a2 = cosv;
    float hid[NBF];
#pragma unroll
    for (int j = 0; j < NBF; ++j)
      hid[j] = silu_f(a0 * Wa1[j] + a1 * Wa1[NBF + j] + a2 * Wa1[2 * NBF + j] + ba1[j]);
#pragma unroll
    for (int o = 0; o < NBF; ++o) {
      float s = ba2[o];
#pragma unroll
      for (int j = 0; j < NBF; ++j) s += hid[j] * Wa2[j * NBF + o];
      af[o] = s;
    }
  } else {
#pragma unroll
    for (int o = 0; o < NBF; ++o) af[o] = 0.0f;
  }
#pragma unroll
  for (int o = 0; o < NBF; ++o) As[r][256 + o] = (__bf16)af[o];
#pragma unroll
  for (int c = KIN; c < KPAD; ++c) As[r][c] = (__bf16)0.0f;
}

// GEMM1 (silu(A@W1+b1) -> Hs) then GEMM2 (Hs@W2+b2 -> atomic scatter).
static __device__ __forceinline__ void gemms_and_scatter(
    __bf16 (&As)[16][AS_STRIDE], __bf16 (&Hs)[16][HS_STRIDE], const int* es,
    const __bf16* __restrict__ w1p, const float* __restrict__ b1,
    const __bf16* __restrict__ w2p, const float* __restrict__ b2,
    float* __restrict__ delta, int lane, int wave) {
  const int g  = lane >> 4;
  const int nl = lane & 15;
  const int nt0 = wave, nt1 = wave + 4;
  {
    v8f acc0 = {}, acc1 = {};
#pragma unroll
    for (int ks = 0; ks < KPAD / 32; ++ks) {
      v16bf a  = load_afrag(&As[nl][0], ks * 32, g);
      v16bf b0 = load_bfrag(w1p, ks * 8 + nt0, lane);
      v16bf bx = load_bfrag(w1p, ks * 8 + nt1, lane);
      if (ks + 1 < KPAD / 32) {
        __builtin_prefetch(w1p + ((ks + 1) * 8 + nt0) * 512 + lane * 16, 0, 3);
        __builtin_prefetch(w1p + ((ks + 1) * 8 + nt1) * 512 + lane * 16, 0, 3);
      }
      acc0 = __builtin_amdgcn_wmma_f32_16x16x32_bf16(false, a, false, b0, (short)0, acc0, false, false);
      acc1 = __builtin_amdgcn_wmma_f32_16x16x32_bf16(false, a, false, bx, (short)0, acc1, false, false);
    }
    int n0 = nt0 * 16 + nl, n1 = nt1 * 16 + nl;
    float bb0 = b1[n0], bb1 = b1[n1];
#pragma unroll
    for (int v = 0; v < 8; ++v) {
      int mm = v + 8 * g;     // C/D: lane -> N, vgpr v -> M = v + 8*(lane>=16)
      Hs[mm][n0] = (__bf16)silu_f(acc0[v] + bb0);
      Hs[mm][n1] = (__bf16)silu_f(acc1[v] + bb1);
    }
  }
  __syncthreads();
  {
    v8f acc0 = {}, acc1 = {};
#pragma unroll
    for (int ks = 0; ks < DMODEL / 32; ++ks) {
      v16bf a  = load_afrag(&Hs[nl][0], ks * 32, g);
      v16bf b0 = load_bfrag(w2p, ks * 8 + nt0, lane);
      v16bf bx = load_bfrag(w2p, ks * 8 + nt1, lane);
      acc0 = __builtin_amdgcn_wmma_f32_16x16x32_bf16(false, a, false, b0, (short)0, acc0, false, false);
      acc1 = __builtin_amdgcn_wmma_f32_16x16x32_bf16(false, a, false, bx, (short)0, acc1, false, false);
    }
    int n0 = nt0 * 16 + nl, n1 = nt1 * 16 + nl;
    float bb0 = b2[n0], bb1 = b2[n1];
#pragma unroll
    for (int v = 0; v < 8; ++v) {
      int mm = v + 8 * g;
      int e = es[mm];
      if (e >= 0) {
        __hip_atomic_fetch_add(&delta[(long long)e * DMODEL + n0], acc0[v] + bb0,
                               __ATOMIC_RELAXED, __HIP_MEMORY_SCOPE_AGENT);
        __hip_atomic_fetch_add(&delta[(long long)e * DMODEL + n1], acc1[v] + bb1,
                               __ATOMIC_RELAXED, __HIP_MEMORY_SCOPE_AGENT);
      }
    }
  }
}

// ----------------- stage 1, TDM path (bf16 edge_attr cache) ----------------

__global__ __launch_bounds__(128) void triplet_tdm_kernel(
    const __bf16* __restrict__ eab,        // E x 128 bf16 cache
    const int*    __restrict__ tbe,        // T x 2
    const float4* __restrict__ geom,
    const float* __restrict__ Wa1, const float* __restrict__ ba1,
    const float* __restrict__ Wa2, const float* __restrict__ ba2,
    const __bf16* __restrict__ w1p, const float* __restrict__ b1,
    const __bf16* __restrict__ w2p, const float* __restrict__ b2,
    float* __restrict__ delta, long long T, int E) {
  __shared__ alignas(16) __bf16 As[16][AS_STRIDE];
  __shared__ alignas(16) __bf16 Hs[16][HS_STRIDE];
  __shared__ int es[16];
  __shared__ int ek[16];

  const int tid  = threadIdx.x;
  const int lane = tid & 31;
  const int wave = tid >> 5;
  const long long tile = blockIdx.x;

  if (tid < 16) {
    long long t = tile * 16 + tid;
    if (t < T) { es[tid] = tbe[2 * t]; ek[tid] = tbe[2 * t + 1]; }
    else       { es[tid] = -1;         ek[tid] = 0; }
  }
  __syncthreads();

  // Each wave issues one 8-row TDM gather (EXEC-independent, one per wave):
  //   wave0: e_ij rows 0-7   -> As[r][0..127]
  //   wave1: e_ij rows 8-15
  //   wave2: e_ik rows 0-7   -> As[r][128..255]
  //   wave3: e_ik rows 8-15
  {
    const int* src = (wave & 2) ? ek : es;
    const int r0   = (wave & 1) * 8;
    const int cb   = (wave & 2) ? 128 : 0;
    int idx[8];
#pragma unroll
    for (int i = 0; i < 8; ++i) {
      int e = __builtin_amdgcn_readfirstlane(src[r0 + i]);
      idx[i] = (e >= 0) ? e : 0x40000000;   // OOB rows gather zeros
    }
    unsigned lds0 = (unsigned)(unsigned long long)(&As[r0][cb]);
    tdm_gather8(lds0, eab, idx, (unsigned)E);
  }

  // Angle MLP overlaps the DMA (disjoint LDS bytes 512..575 of each row).
  if (tid < 16) angle_row(tid, es, ek, geom, Wa1, ba1, Wa2, ba2, As);

  wait_tensorcnt0();
  __syncthreads();

  gemms_and_scatter(As, Hs, es, w1p, b1, w2p, b2, delta, lane, wave);
}

// -------------- stage 1, fallback path (synchronous gather) ----------------

__global__ __launch_bounds__(128) void triplet_gather_kernel(
    const float* __restrict__ ea,
    const int*   __restrict__ tbe,
    const float4* __restrict__ geom,
    const float* __restrict__ Wa1, const float* __restrict__ ba1,
    const float* __restrict__ Wa2, const float* __restrict__ ba2,
    const __bf16* __restrict__ w1p, const float* __restrict__ b1,
    const __bf16* __restrict__ w2p, const float* __restrict__ b2,
    float* __restrict__ delta, long long T) {
  __shared__ alignas(16) __bf16 As[16][AS_STRIDE];
  __shared__ alignas(16) __bf16 Hs[16][HS_STRIDE];
  __shared__ int es[16];
  __shared__ int ek[16];

  const int tid  = threadIdx.x;
  const int lane = tid & 31;
  const int wave = tid >> 5;
  const long long tile = blockIdx.x;

  if (tid < 16) {
    long long t = tile * 16 + tid;
    if (t < T) { es[tid] = tbe[2 * t]; ek[tid] = tbe[2 * t + 1]; }
    else       { es[tid] = -1;         ek[tid] = 0; }
  }
  __syncthreads();

  for (int i = tid; i < 16 * 256; i += 128) {
    int r = i >> 8, c = i & 255;
    float v = 0.0f;
    if (es[r] >= 0)
      v = (c < 128) ? ea[(long long)es[r] * DMODEL + c]
                    : ea[(long long)ek[r] * DMODEL + (c - 128)];
    As[r][c] = (__bf16)v;
  }
  if (tid < 16) angle_row(tid, es, ek, geom, Wa1, ba1, Wa2, ba2, As);
  __syncthreads();

  gemms_and_scatter(As, Hs, es, w1p, b1, w2p, b2, delta, lane, wave);
}

// ------------------------------ stage 2 ------------------------------------

__global__ __launch_bounds__(128) void edge_out_kernel(
    float* __restrict__ out, const __bf16* __restrict__ wup,
    const float* __restrict__ bu, int E) {
  __shared__ alignas(16) __bf16 Ds[16][HS_STRIDE];

  const int tid  = threadIdx.x;
  const int lane = tid & 31;
  const int wave = tid >> 5;
  const int g    = lane >> 4;
  const int nl   = lane & 15;
  const int base = blockIdx.x * 16;

  for (int i = tid; i < 16 * 128; i += 128) {
    int r = i >> 7, c = i & 127;
    int row = base + r;
    float v = (row < E) ? out[(long long)row * DMODEL + c] : 0.0f;
    Ds[r][c] = (__bf16)v;
  }
  __syncthreads();

  const int nt0 = wave, nt1 = wave + 4;
  v8f acc0 = {}, acc1 = {};
#pragma unroll
  for (int ks = 0; ks < DMODEL / 32; ++ks) {
    v16bf a  = load_afrag(&Ds[nl][0], ks * 32, g);
    v16bf b0 = load_bfrag(wup, ks * 8 + nt0, lane);
    v16bf bx = load_bfrag(wup, ks * 8 + nt1, lane);
    acc0 = __builtin_amdgcn_wmma_f32_16x16x32_bf16(false, a, false, b0, (short)0, acc0, false, false);
    acc1 = __builtin_amdgcn_wmma_f32_16x16x32_bf16(false, a, false, bx, (short)0, acc1, false, false);
  }
  int n0 = nt0 * 16 + nl, n1 = nt1 * 16 + nl;
  float bb0 = bu[n0], bb1 = bu[n1];
#pragma unroll
  for (int v = 0; v < 8; ++v) {
    int mm = v + 8 * g;
    int row = base + mm;
    if (row < E) {
      float v0 = acc0[v] + bb0;
      float v1 = acc1[v] + bb1;
      if (!__builtin_isfinite(v0)) v0 = 0.0f;
      if (!__builtin_isfinite(v1)) v1 = 0.0f;
      out[(long long)row * DMODEL + n0] = v0;
      out[(long long)row * DMODEL + n1] = v1;
    }
  }
}

// ------------------------------ launcher -----------------------------------

extern "C" void kernel_launch(void* const* d_in, const int* in_sizes, int n_in,
                              void* d_out, int out_size, void* d_ws, size_t ws_size,
                              hipStream_t stream) {
  const float* ea  = (const float*)d_in[0];
  const int*   tbe = (const int*)d_in[2];
  const float* ev  = (const float*)d_in[3];
  const float* Wa1 = (const float*)d_in[4];
  const float* ba1 = (const float*)d_in[5];
  const float* Wa2 = (const float*)d_in[6];
  const float* ba2 = (const float*)d_in[7];
  const float* W1  = (const float*)d_in[8];
  const float* b1  = (const float*)d_in[9];
  const float* W2  = (const float*)d_in[10];
  const float* b2  = (const float*)d_in[11];
  const float* Wu  = (const float*)d_in[12];
  const float* bu  = (const float*)d_in[13];

  const int       E = in_sizes[0] / DMODEL;
  const long long T = (long long)in_sizes[2] / 2;
  float* outp = (float*)d_out;

  size_t off = 0;
  auto carve = [&](size_t bytes) {
    void* p = (char*)d_ws + off;
    off += (bytes + 255) & ~(size_t)255;
    return p;
  };
  float4* geom = (float4*)carve((size_t)E * sizeof(float4));
  __bf16* w1p  = (__bf16*)carve((size_t)(KPAD / 32) * 8 * 512 * sizeof(__bf16));
  __bf16* w2p  = (__bf16*)carve((size_t)(DMODEL / 32) * 8 * 512 * sizeof(__bf16));
  __bf16* wup  = (__bf16*)carve((size_t)(DMODEL / 32) * 8 * 512 * sizeof(__bf16));
  size_t base_need = off;
  __bf16* eab  = (__bf16*)carve((size_t)E * DMODEL * sizeof(__bf16));
  const bool use_tdm = (ws_size >= off) && (base_need <= ws_size);

  zero_f32_kernel<<<1024, 256, 0, stream>>>(outp, (long long)E * DMODEL);
  geom_kernel<<<(E + 255) / 256, 256, 0, stream>>>(ev, geom, E);

  const int n1 = (KPAD / 32) * 8 * 512;
  const int n2 = (DMODEL / 32) * 8 * 512;
  pack_b_kernel<<<(n1 + 255) / 256, 256, 0, stream>>>(W1, w1p, KIN, KPAD);
  pack_b_kernel<<<(n2 + 255) / 256, 256, 0, stream>>>(W2, w2p, DMODEL, DMODEL);
  pack_b_kernel<<<(n2 + 255) / 256, 256, 0, stream>>>(Wu, wup, DMODEL, DMODEL);

  const unsigned ntiles = (unsigned)((T + 15) / 16);
  if (use_tdm) {
    cvt_bf16_kernel<<<2048, 256, 0, stream>>>(ea, eab, (long long)E * DMODEL);
    triplet_tdm_kernel<<<ntiles, 128, 0, stream>>>(eab, tbe, geom, Wa1, ba1, Wa2, ba2,
                                                   w1p, b1, w2p, b2, outp, T, E);
  } else {
    triplet_gather_kernel<<<ntiles, 128, 0, stream>>>(ea, tbe, geom, Wa1, ba1, Wa2, ba2,
                                                      w1p, b1, w2p, b2, outp, T);
  }

  edge_out_kernel<<<(E + 15) / 16, 128, 0, stream>>>(outp, wup, bu, E);
}
